// VectorQuantization_5068061409926
// MI455X (gfx1250) — matrix operand order repaired and checked by reading, not compiled, not run on hardware
//
#include <hip/hip_runtime.h>
#include <stdint.h>

// Problem constants (match reference)
#define B_ 8
#define D_ 512
#define N_ 4096
#define K_ 2048

typedef __attribute__((ext_vector_type(16))) __bf16        v16bf;
typedef __attribute__((ext_vector_type(8)))  float         v8f;
typedef __attribute__((ext_vector_type(4)))  unsigned int  u32x4;

union FragU { u32x4 q[2]; v16bf b; };

// Split fp32 into bf16 hi + bf16 lo (truncating): x ~= hi + lo with ~2^-16 rel err.
__device__ __forceinline__ void split_bf16(float x, unsigned short& hi, unsigned short& lo) {
  union { float f; unsigned int u; } c; c.f = x;
  unsigned int hb = c.u & 0xFFFF0000u;
  hi = (unsigned short)(hb >> 16);
  union { unsigned int u; float f; } h; h.u = hb;
  union { float f; unsigned int u; } r; r.f = x - h.f;
  lo = (unsigned short)(r.u >> 16);
}

// Async copy: 16 bytes per lane, global -> LDS, tracked by ASYNCcnt.
// lds  = LDS byte offset (low 32 bits of generic shared pointer)
// gptr = per-lane 64-bit global byte address
__device__ __forceinline__ void async_b128(unsigned lds, unsigned long long gptr) {
  asm volatile("global_load_async_to_lds_b128 %0, %1, off"
               :: "v"(lds), "v"(gptr) : "memory");
}

// ---- Phase 0a: x [B][D][N] fp32 -> xhi/xlo bf16 [B*N][D] (transpose via LDS tile)
__global__ void k_convert_x(const float* __restrict__ x,
                            unsigned short* __restrict__ xhi,
                            unsigned short* __restrict__ xlo) {
  __shared__ float t[32][33];
  const int b  = blockIdx.z;
  const int d0 = blockIdx.y * 32;
  const int n0 = blockIdx.x * 32;
  const int tx = threadIdx.x & 31;
  const int ty = threadIdx.x >> 5;            // 0..7
  const float* xb = x + (size_t)b * D_ * N_;
#pragma unroll
  for (int j = 0; j < 4; ++j) {
    int d = d0 + ty + 8 * j;
    t[ty + 8 * j][tx] = xb[(size_t)d * N_ + n0 + tx];   // coalesced along n
  }
  __syncthreads();
#pragma unroll
  for (int j = 0; j < 4; ++j) {
    int n = n0 + ty + 8 * j;
    float v = t[tx][ty + 8 * j];
    unsigned short h, l; split_bf16(v, h, l);
    size_t o = ((size_t)b * N_ + n) * D_ + d0 + tx;     // coalesced along d
    xhi[o] = h; xlo[o] = l;
  }
}

// ---- Phase 0b: embed [K][D] fp32 -> ehi/elo bf16 [K][D], plus e2[k] = sum_d e^2
__global__ void k_convert_e(const float* __restrict__ e,
                            unsigned short* __restrict__ ehi,
                            unsigned short* __restrict__ elo,
                            float* __restrict__ e2) {
  const int row  = (int)((blockIdx.x * blockDim.x + threadIdx.x) >> 5);  // one wave32 per k-row
  const int lane = threadIdx.x & 31;
  if (row >= K_) return;
  const float* src = e + (size_t)row * D_;
  float s = 0.f;
#pragma unroll
  for (int i = 0; i < D_ / 32; ++i) {
    float v = src[lane + 32 * i];
    s += v * v;
    unsigned short h, l; split_bf16(v, h, l);
    size_t o = (size_t)row * D_ + lane + 32 * i;
    ehi[o] = h; elo[o] = l;
  }
#pragma unroll
  for (int off = 16; off >= 1; off >>= 1) s += __shfl_xor(s, off, 32);
  if (lane == 0) e2[row] = s;
}

// ---- Phase 1: fused WMMA cross-term + argmin, async double-buffered staging
// Block: 128 threads (4 waves), owns 64 query rows (16 per wave).
// k outer (chunks of 64 -> 4 accumulator subtiles), d inner (chunks of 32, 2-deep pipeline).
#define LDSPITCH 40                       // elements; 80 B rows: 16B-aligned b128, 20-bank stride
#define TILE_ELEMS (64 * LDSPITCH)        // one buffer of one array
#define BUFBYTES (TILE_ELEMS * 2)         // 5120 B

__global__ void __launch_bounds__(128)
k_argmin(const unsigned short* __restrict__ xhi,
         const unsigned short* __restrict__ xlo,
         const unsigned short* __restrict__ ehi,
         const unsigned short* __restrict__ elo,
         const float* __restrict__ e2,
         int* __restrict__ ind) {
  __shared__ __align__(16) unsigned short sXh[2][TILE_ELEMS];
  __shared__ __align__(16) unsigned short sXl[2][TILE_ELEMS];
  __shared__ __align__(16) unsigned short sEh[2][TILE_ELEMS];
  __shared__ __align__(16) unsigned short sEl[2][TILE_ELEMS];

  const int tid  = threadIdx.x;
  const int lane = tid & 31;
  const int wave = tid >> 5;                 // 0..3
  const int rowBaseX = blockIdx.x * 64;      // flat (b*N + n) base

  // ---- Staging geometry: thread -> (row0 = tid/4 in 0..31, seg = tid%4), plus row0+32.
  const int row0 = tid >> 2;
  const int seg  = tid & 3;

  // LDS byte offsets (buffer 0); +BUFBYTES selects buffer 1, +ldsRow32 selects rows 32..63.
  const unsigned lXh = (unsigned)(uintptr_t)&sXh[0][row0 * LDSPITCH + seg * 8];
  const unsigned lXl = (unsigned)(uintptr_t)&sXl[0][row0 * LDSPITCH + seg * 8];
  const unsigned lEh = (unsigned)(uintptr_t)&sEh[0][row0 * LDSPITCH + seg * 8];
  const unsigned lEl = (unsigned)(uintptr_t)&sEl[0][row0 * LDSPITCH + seg * 8];
  const unsigned ldsRow32 = 32u * LDSPITCH * 2u;            // 2560 B

  // Global byte bases for x (k-invariant). d-chunk advance = 64 B; rows +32 = +32 KB.
  const unsigned long long gXh =
      (unsigned long long)(uintptr_t)(xhi + ((size_t)(rowBaseX + row0)) * D_ + seg * 8);
  const unsigned long long gXl =
      (unsigned long long)(uintptr_t)(xlo + ((size_t)(rowBaseX + row0)) * D_ + seg * 8);
  const unsigned long long gRow32 = (unsigned long long)32 * D_ * 2;  // 32768 B

  // ---- Fragment addressing (ISA 7.12.2)
  const int half = lane >> 4;                // 0: lanes 0-15, 1: lanes 16-31
  const int l15  = lane & 15;
  const int aRow = 16 * wave + l15;          // A: lane<16 -> K{0..7,16..23}; else K{8..15,24..31}
  const int aC0  = half ? 8 : 0;
  const int bC0  = half ? 8 : 0;             // B: lane<16 -> K{0..7,16..23}; else K{8..15,24..31}

  float best[8];
  int   bidx[8];
#pragma unroll
  for (int r = 0; r < 8; ++r) { best[r] = 3.4e38f; bidx[r] = 0; }

  for (int k0 = 0; k0 < K_; k0 += 64) {
    const unsigned long long gEh =
        (unsigned long long)(uintptr_t)(ehi + ((size_t)(k0 + row0)) * D_ + seg * 8);
    const unsigned long long gEl =
        (unsigned long long)(uintptr_t)(elo + ((size_t)(k0 + row0)) * D_ + seg * 8);

    // Issue one full stage (8 async b128 per thread) for d-chunk at byte offset dB into buffer buf.
    auto issue = [&](int buf, unsigned long long dB) {
      const unsigned o = buf ? (unsigned)BUFBYTES : 0u;
      async_b128(lXh + o,            gXh + dB);
      async_b128(lXh + o + ldsRow32, gXh + gRow32 + dB);
      async_b128(lXl + o,            gXl + dB);
      async_b128(lXl + o + ldsRow32, gXl + gRow32 + dB);
      async_b128(lEh + o,            gEh + dB);
      async_b128(lEh + o + ldsRow32, gEh + gRow32 + dB);
      async_b128(lEl + o,            gEl + dB);
      async_b128(lEl + o + ldsRow32, gEl + gRow32 + dB);
    };

    // Prologue: fill both pipeline stages (d-chunks 0 and 1).
    issue(0, 0);
    issue(1, 64);

    v8f acc[4] = {};   // 4 subtiles of 16 codes each

    for (int ic = 0; ic < 16; ++ic) {
      const int cur = ic & 1;
      // Retire stage ic: 8 per stage, loads complete in order; while draining, wait to 0.
      if (ic < 14) asm volatile("s_wait_asynccnt 0x8" ::: "memory");
      else         asm volatile("s_wait_asynccnt 0x0" ::: "memory");
      __syncthreads();   // stage ic visible block-wide

      const unsigned short* pXh = sXh[cur];
      const unsigned short* pXl = sXl[cur];
      const unsigned short* pEh = sEh[cur];
      const unsigned short* pEl = sEl[cur];

      FragU ah, al;
      ah.q[0] = *(const u32x4*)(pXh + aRow * LDSPITCH + aC0);
      ah.q[1] = *(const u32x4*)(pXh + aRow * LDSPITCH + aC0 + 16);
      al.q[0] = *(const u32x4*)(pXl + aRow * LDSPITCH + aC0);
      al.q[1] = *(const u32x4*)(pXl + aRow * LDSPITCH + aC0 + 16);

#pragma unroll
      for (int s = 0; s < 4; ++s) {
        const int bRow = 16 * s + l15;
        FragU bh, bl;
        bh.q[0] = *(const u32x4*)(pEh + bRow * LDSPITCH + bC0);
        bh.q[1] = *(const u32x4*)(pEh + bRow * LDSPITCH + bC0 + 16);
        bl.q[0] = *(const u32x4*)(pEl + bRow * LDSPITCH + bC0);
        bl.q[1] = *(const u32x4*)(pEl + bRow * LDSPITCH + bC0 + 16);
        // 3-term split-bf16 fp32 emulation: hi*hi + hi*lo + lo*hi
        acc[s] = __builtin_amdgcn_wmma_f32_16x16x32_bf16(false, ah.b, false, bh.b, (short)0, acc[s], false, false);
        acc[s] = __builtin_amdgcn_wmma_f32_16x16x32_bf16(false, ah.b, false, bl.b, (short)0, acc[s], false, false);
        acc[s] = __builtin_amdgcn_wmma_f32_16x16x32_bf16(false, al.b, false, bh.b, (short)0, acc[s], false, false);
      }

      __syncthreads();   // all waves done reading buffer cur
      if (ic + 2 < 16) issue(cur, (unsigned long long)(ic + 2) * 64);
    }

    // Epilogue: dist-equivalent = e2[k] - 2*cross  (x^2 term constant in k, dropped)
    // C layout: VGPR r, lane<16 -> (M=r, N=l15); lane>=16 -> (M=r+8, N=l15)
#pragma unroll
    for (int s = 0; s < 4; ++s) {
      int k = k0 + 16 * s + l15;
      float ek = e2[k];
#pragma unroll
      for (int r = 0; r < 8; ++r) {
        float d2 = ek - 2.0f * acc[s][r];
        if (d2 < best[r] || (d2 == best[r] && k < bidx[r])) { best[r] = d2; bidx[r] = k; }
      }
    }
  }

  // Reduce argmin across the 16 lanes of each half (same M per half)
#pragma unroll
  for (int r = 0; r < 8; ++r) {
    float bv = best[r]; int iv = bidx[r];
#pragma unroll
    for (int off = 8; off >= 1; off >>= 1) {
      float ob = __shfl_xor(bv, off, 32);
      int   oi = __shfl_xor(iv, off, 32);
      if (ob < bv || (ob == bv && oi < iv)) { bv = ob; iv = oi; }
    }
    if (l15 == 0) {
      int m = 16 * wave + 8 * half + r;
      ind[rowBaseX + m] = iv;
    }
  }
}

// ---- Phase 2: gather quantize[b][d][n] = embed[ind[b][n]][d]  (embed lives in L2)
__global__ void k_gather(const float* __restrict__ embed,
                         const int* __restrict__ ind,
                         float* __restrict__ out) {
  const size_t total = (size_t)B_ * D_ * N_;
  for (size_t o = (size_t)blockIdx.x * blockDim.x + threadIdx.x; o < total;
       o += (size_t)gridDim.x * blockDim.x) {
    size_t b = o / ((size_t)D_ * N_);
    size_t rem = o % ((size_t)D_ * N_);
    int d = (int)(rem / N_);
    int n = (int)(rem % N_);
    int k = ind[b * N_ + n];
    out[o] = embed[(size_t)k * D_ + d];
  }
}

__global__ void k_write_ind(const int* __restrict__ ind, float* __restrict__ out) {
  int i = blockIdx.x * blockDim.x + threadIdx.x;
  if (i < B_ * N_) out[i] = (float)ind[i];
}

extern "C" void kernel_launch(void* const* d_in, const int* in_sizes, int n_in,
                              void* d_out, int out_size, void* d_ws, size_t ws_size,
                              hipStream_t stream) {
  const float* x     = (const float*)d_in[0];   // [B, D, N] fp32
  const float* embed = (const float*)d_in[1];   // [K, D]    fp32
  float* out = (float*)d_out;                   // [B*D*N] quantize fp32, then [B*N] indices as fp32

  // Workspace carve-up (~71.3 MB total)
  char* ws = (char*)d_ws;
  size_t off = 0;
  unsigned short* xhi = (unsigned short*)(ws + off); off += (size_t)B_ * N_ * D_ * 2;  // 32 MB
  unsigned short* xlo = (unsigned short*)(ws + off); off += (size_t)B_ * N_ * D_ * 2;  // 32 MB
  unsigned short* ehi = (unsigned short*)(ws + off); off += (size_t)K_ * D_ * 2;       // 2 MB
  unsigned short* elo = (unsigned short*)(ws + off); off += (size_t)K_ * D_ * 2;       // 2 MB
  float*          e2w = (float*)(ws + off);          off += (size_t)K_ * 4;            // 8 KB
  int*            ind = (int*)(ws + off);            off += (size_t)B_ * N_ * 4;       // 128 KB

  dim3 gx(N_ / 32, D_ / 32, B_);
  k_convert_x<<<gx, 256, 0, stream>>>(x, xhi, xlo);
  k_convert_e<<<(K_ * 32) / 256, 256, 0, stream>>>(embed, ehi, elo, e2w);
  k_argmin<<<(B_ * N_) / 64, 128, 0, stream>>>(xhi, xlo, ehi, elo, e2w, ind);
  k_gather<<<2048, 256, 0, stream>>>(embed, ind, out);
  k_write_ind<<<(B_ * N_ + 255) / 256, 256, 0, stream>>>(ind, out + (size_t)B_ * D_ * N_);
}
